// Rwkv5SelfAttention_8744553415392
// MI455X (gfx1250) — compile-verified, matrix operand order checked by reading, not run on hardware
//
#include <hip/hip_runtime.h>
#include <hip/hip_bf16.h>
#include <math.h>

// ---------------------------------------------------------------------------
// RWKV5 self-attention forward, MI455X (gfx1250), wave32 + WMMA bf16.
//   B=4, T=2048, C=2048, H=32, S=64
// Pipeline:
//   1. f32->bf16 convert of the 5 CxC weights
//   2. token-shift mix -> 4 bf16 activations
//   3. 4x WMMA GEMM (k, v, r, silu(gate))      [compute bound, bf16 WMMA]
//   4. recurrent scan, register-resident state, double-buffered
//      async-to-LDS prefetch of r/k/v (ASYNCcnt path)        [latency bound]
//   5. per-(token,head) groupnorm * ln_w + ln_b, * gate
//   6. final WMMA GEMM -> f32 output; scan already wrote state_f
// ---------------------------------------------------------------------------

#define Bc 4
#define Tc 2048
#define Cc 2048
#define Hc 32
#define Sc 64
#define Mrows (Bc * Tc)            // 8192

typedef __bf16 v16bf __attribute__((ext_vector_type(16)));
typedef float  v8f   __attribute__((ext_vector_type(8)));
typedef unsigned int u32x4 __attribute__((ext_vector_type(4)));

union Frag {
  u32x4 q[2];
  v16bf v;
};

#if __has_builtin(__builtin_amdgcn_global_load_async_to_lds_b128) && \
    __has_builtin(__builtin_amdgcn_s_wait_asynccnt)
#define HAVE_ASYNC_LDS 1
// Builtin parameter types (from hipcc diagnostics):
//   param 1: int __attribute__((vector_size(16))) addrspace(1)*  (global src)
//   param 2: LDS-side pointer of the same pointee type
typedef int v4i __attribute__((vector_size(16)));
typedef __attribute__((address_space(1))) v4i* gv4i_p;
typedef __attribute__((address_space(3))) v4i* lv4i_p;
#else
#define HAVE_ASYNC_LDS 0
#endif

// ------------------------------ weight convert -----------------------------
__global__ __launch_bounds__(256) void f32_to_bf16_kernel(
    const float* __restrict__ src, __bf16* __restrict__ dst, int n) {
  int i = blockIdx.x * 256 + threadIdx.x;
  if (i < n) dst[i] = (__bf16)src[i];
}

// ------------------------------ token-shift mix ----------------------------
__global__ __launch_bounds__(256) void mix_kernel(
    const float* __restrict__ hidden,
    const float* __restrict__ mk, const float* __restrict__ mv,
    const float* __restrict__ mr, const float* __restrict__ mg,
    __bf16* __restrict__ xk, __bf16* __restrict__ xv,
    __bf16* __restrict__ xr, __bf16* __restrict__ xg) {
  size_t idx = (size_t)blockIdx.x * 256 + threadIdx.x;   // over B*T*C
  int c = (int)(idx & (Cc - 1));
  size_t m = idx >> 11;                                   // token row
  int t = (int)(m & (Tc - 1));
  float h0 = hidden[idx];
  float hs = (t == 0) ? 0.f : hidden[idx - Cc];
  float a;
  a = mk[c]; xk[idx] = (__bf16)(h0 * a + hs * (1.f - a));
  a = mv[c]; xv[idx] = (__bf16)(h0 * a + hs * (1.f - a));
  a = mr[c]; xr[idx] = (__bf16)(h0 * a + hs * (1.f - a));
  a = mg[c]; xg[idx] = (__bf16)(h0 * a + hs * (1.f - a));
}

// ------------------------------ WMMA GEMM ----------------------------------
// Y[M,N] = A[M,K] (bf16, row major) @ W[N,K]^T (bf16, row major)
// MODE 0: store bf16   MODE 1: silu then bf16   MODE 2: store f32
// Block: 256 threads = 8 waves (2x4); wave tile 64x64; block tile 128x256.
template <int MODE>
__global__ __launch_bounds__(256) void gemm_bf16_kernel(
    const __bf16* __restrict__ A, const __bf16* __restrict__ W,
    void* __restrict__ outp, int M, int N, int K) {
  const int lane = threadIdx.x & 31;
  const int wid  = threadIdx.x >> 5;
  const int wm = wid >> 2;            // 0..1
  const int wn = wid & 3;             // 0..3
  const int m_base = blockIdx.x * 128 + wm * 64;
  const int n_base = blockIdx.y * 256 + wn * 64;
  const int row  = lane & 15;
  const int hi   = lane >> 4;         // lane half selects K sub-block
  const int koff = hi * 8;

  v8f acc[4][4];
#pragma unroll
  for (int i = 0; i < 4; ++i)
#pragma unroll
    for (int j = 0; j < 4; ++j) acc[i][j] = (v8f)0.f;

  for (int k = 0; k < K; k += 32) {
    Frag a[4], b[4];
#pragma unroll
    for (int i = 0; i < 4; ++i) {
      const __bf16* pa = A + (size_t)(m_base + i * 16 + row) * K + k + koff;
      a[i].q[0] = *(const u32x4*)(pa);
      a[i].q[1] = *(const u32x4*)(pa + 16);
      const __bf16* pb = W + (size_t)(n_base + i * 16 + row) * K + k + koff;
      b[i].q[0] = *(const u32x4*)(pb);
      b[i].q[1] = *(const u32x4*)(pb + 16);
    }
#pragma unroll
    for (int i = 0; i < 4; ++i)
#pragma unroll
      for (int j = 0; j < 4; ++j)
        acc[i][j] = __builtin_amdgcn_wmma_f32_16x16x32_bf16(
            false, a[i].v, false, b[j].v, (short)0, acc[i][j], false, false);
  }

#pragma unroll
  for (int i = 0; i < 4; ++i) {
#pragma unroll
    for (int j = 0; j < 4; ++j) {
      int n = n_base + j * 16 + (lane & 15);
#pragma unroll
      for (int vv = 0; vv < 8; ++vv) {
        int m = m_base + i * 16 + hi * 8 + vv;
        float val = acc[i][j][vv];
        if (MODE == 0) {
          ((__bf16*)outp)[(size_t)m * N + n] = (__bf16)val;
        } else if (MODE == 1) {
          float s = val / (1.f + __expf(-val));   // silu
          ((__bf16*)outp)[(size_t)m * N + n] = (__bf16)s;
        } else {
          ((float*)outp)[(size_t)m * N + n] = val;
        }
      }
    }
  }
}

// ------------------------------ recurrent scan -----------------------------
// One block per (b,h). 256 threads: d = tid&63 (state column), sg = tid>>6
// (16-row slice of s). State lives in registers: 16 floats per thread.
// r/k/v for step t+1 are prefetched into the alternate LDS stage with
// global_load_async_to_lds_b128 while step t computes (ASYNCcnt tracked,
// in-order completion => s_wait_asynccnt 1 guarantees current stage ready).
__global__ __launch_bounds__(256) void rwkv_scan_kernel(
    const __bf16* __restrict__ rb, const __bf16* __restrict__ kb,
    const __bf16* __restrict__ vb, const float* __restrict__ state0,
    const float* __restrict__ tdecay, const float* __restrict__ tfaaaa,
    __bf16* __restrict__ xout, float* __restrict__ state_out) {
  const int bh = blockIdx.x;
  const int b = bh / Hc;
  const int h = bh % Hc;
  const int tid = threadIdx.x;
  const int d  = tid & 63;
  const int sg = tid >> 6;            // 0..3
  const int s0 = sg * 16;

  __shared__ __attribute__((aligned(16))) __bf16 stage[2][3][Sc];
  __shared__ float part[256];

  float st[16], wreg[16], ureg[16];
#pragma unroll
  for (int i = 0; i < 16; ++i) {
    int s = s0 + i;
    st[i]   = state0[(((size_t)b * Hc + h) * Sc + s) * Sc + d];
    wreg[i] = __expf(-__expf(tdecay[h * Sc + s]));
    ureg[i] = tfaaaa[h * Sc + s];
  }

#if HAVE_ASYNC_LDS
  // Issue one 128-bit async DMA per (array, 16B-segment): 24 lanes cover
  // the 3 x 64 bf16 (3 x 128B) stage for one timestep.
  {
    if (tid < 24) {
      int arr = tid >> 3;
      int seg = tid & 7;
      size_t gbase = (size_t)(b * Tc + 0) * Cc + h * Sc + seg * 8;
      const __bf16* src = (arr == 0 ? rb : arr == 1 ? kb : vb) + gbase;
      __builtin_amdgcn_global_load_async_to_lds_b128(
          (gv4i_p)src, (lv4i_p)&stage[0][arr][seg * 8], 0, 0);
    }
  }
#endif

  for (int t = 0; t < Tc; ++t) {
    const int buf = t & 1;
    size_t base = (size_t)(b * Tc + t) * Cc + h * Sc;

#if HAVE_ASYNC_LDS
    if (t + 1 < Tc) {
      if (tid < 24) {
        int arr = tid >> 3;
        int seg = tid & 7;
        const __bf16* src = (arr == 0 ? rb : arr == 1 ? kb : vb) +
                            (base + Cc + seg * 8);
        __builtin_amdgcn_global_load_async_to_lds_b128(
            (gv4i_p)src, (lv4i_p)&stage[buf ^ 1][arr][seg * 8], 0, 0);
      }
      __builtin_amdgcn_s_wait_asynccnt(1);  // current stage complete (in-order)
    } else {
      __builtin_amdgcn_s_wait_asynccnt(0);
    }
    __syncthreads();                        // stage[buf] visible to all waves
#else
    __syncthreads();                        // protect stage reuse
    if (tid < Sc) {
      stage[buf][0][tid] = rb[base + tid];
      stage[buf][1][tid] = kb[base + tid];
      stage[buf][2][tid] = vb[base + tid];
    }
    __syncthreads();
#endif

    float vd = (float)stage[buf][2][d];
    float o = 0.f;
#pragma unroll
    for (int i = 0; i < 16; ++i) {
      int s = s0 + i;
      float kv = (float)stage[buf][1][s] * vd;
      o = fmaf((float)stage[buf][0][s], fmaf(ureg[i], kv, st[i]), o);
      st[i] = fmaf(wreg[i], st[i], kv);
    }
    part[tid] = o;
    __syncthreads();                        // also fences stage[buf] reads
    if (sg == 0) {
      float tot = part[d] + part[d + 64] + part[d + 128] + part[d + 192];
      xout[base + d] = (__bf16)tot;
    }
  }

#pragma unroll
  for (int i = 0; i < 16; ++i) {
    int s = s0 + i;
    state_out[(((size_t)b * Hc + h) * Sc + s) * Sc + d] = st[i];
  }
}

// ------------------------------ groupnorm * gate ---------------------------
__global__ __launch_bounds__(256) void ln_gate_kernel(
    const __bf16* __restrict__ x, const __bf16* __restrict__ gate,
    const float* __restrict__ lnw, const float* __restrict__ lnb,
    __bf16* __restrict__ y) {
  int gid = blockIdx.x * 256 + threadIdx.x;   // over M * H
  int m = gid >> 5;
  int h = gid & 31;
  size_t base = (size_t)m * Cc + h * Sc;
  float xa[Sc];
  float sum = 0.f, sq = 0.f;
#pragma unroll
  for (int s = 0; s < Sc; ++s) {
    float v = (float)x[base + s] * 0.125f;    // / HEAD_SIZE_DIVISOR
    xa[s] = v;
    sum += v;
    sq  += v * v;
  }
  float mean = sum * (1.f / Sc);
  float var  = sq * (1.f / Sc) - mean * mean;
  float rstd = rsqrtf(var + 1e-5f);
#pragma unroll
  for (int s = 0; s < Sc; ++s) {
    float v = (xa[s] - mean) * rstd * lnw[h * Sc + s] + lnb[h * Sc + s];
    v *= (float)gate[base + s];
    y[base + s] = (__bf16)v;
  }
}

// ---------------------------------------------------------------------------
extern "C" void kernel_launch(void* const* d_in, const int* in_sizes, int n_in,
                              void* d_out, int out_size, void* d_ws, size_t ws_size,
                              hipStream_t stream) {
  const float* hidden = (const float*)d_in[0];
  const float* state0 = (const float*)d_in[1];
  const float* tdecay = (const float*)d_in[2];
  const float* tfaaaa = (const float*)d_in[3];
  const float* mk     = (const float*)d_in[4];
  const float* mv     = (const float*)d_in[5];
  const float* mr     = (const float*)d_in[6];
  const float* mg     = (const float*)d_in[7];
  const float* key_w  = (const float*)d_in[8];
  const float* val_w  = (const float*)d_in[9];
  const float* rec_w  = (const float*)d_in[10];
  const float* gate_w = (const float*)d_in[11];
  const float* out_w  = (const float*)d_in[12];
  const float* lnw    = (const float*)d_in[13];
  const float* lnb    = (const float*)d_in[14];

  float* outF   = (float*)d_out;
  float* stateF = outF + (size_t)Bc * Tc * Cc;

  // Workspace layout (bytes). Peak ~296 MiB; x / y alias the dead mix bufs.
  const size_t WB = (size_t)Cc * Cc * 2;       // bf16 weight:       8 MiB
  const size_t AB = (size_t)Mrows * Cc * 2;    // bf16 activation:  32 MiB
  char* ws = (char*)d_ws;
  __bf16* wkb = (__bf16*)(ws + 0 * WB);
  __bf16* wvb = (__bf16*)(ws + 1 * WB);
  __bf16* wrb = (__bf16*)(ws + 2 * WB);
  __bf16* wgb = (__bf16*)(ws + 3 * WB);
  __bf16* wob = (__bf16*)(ws + 4 * WB);
  char*   a0  = ws + 5 * WB;
  __bf16* xk  = (__bf16*)(a0 + 0 * AB);
  __bf16* xv  = (__bf16*)(a0 + 1 * AB);
  __bf16* xr  = (__bf16*)(a0 + 2 * AB);
  __bf16* xg  = (__bf16*)(a0 + 3 * AB);
  __bf16* rbf = (__bf16*)(a0 + 4 * AB);
  __bf16* kbf = (__bf16*)(a0 + 5 * AB);
  __bf16* vbf = (__bf16*)(a0 + 6 * AB);
  __bf16* gbf = (__bf16*)(a0 + 7 * AB);
  __bf16* xbf = xk;                            // alias (mixes dead after GEMMs)
  __bf16* ybf = xv;                            // alias

  // 1) weight conversion
  const int WN = Cc * Cc;
  f32_to_bf16_kernel<<<WN / 256, 256, 0, stream>>>(key_w,  wkb, WN);
  f32_to_bf16_kernel<<<WN / 256, 256, 0, stream>>>(val_w,  wvb, WN);
  f32_to_bf16_kernel<<<WN / 256, 256, 0, stream>>>(rec_w,  wrb, WN);
  f32_to_bf16_kernel<<<WN / 256, 256, 0, stream>>>(gate_w, wgb, WN);
  f32_to_bf16_kernel<<<WN / 256, 256, 0, stream>>>(out_w,  wob, WN);

  // 2) token-shift mixes
  const size_t total = (size_t)Mrows * Cc;
  mix_kernel<<<(unsigned)(total / 256), 256, 0, stream>>>(
      hidden, mk, mv, mr, mg, xk, xv, xr, xg);

  // 3) projection GEMMs
  dim3 ggrid(Mrows / 128, Cc / 256);
  gemm_bf16_kernel<0><<<ggrid, 256, 0, stream>>>(xk, wkb, (void*)kbf, Mrows, Cc, Cc);
  gemm_bf16_kernel<0><<<ggrid, 256, 0, stream>>>(xv, wvb, (void*)vbf, Mrows, Cc, Cc);
  gemm_bf16_kernel<0><<<ggrid, 256, 0, stream>>>(xr, wrb, (void*)rbf, Mrows, Cc, Cc);
  gemm_bf16_kernel<1><<<ggrid, 256, 0, stream>>>(xg, wgb, (void*)gbf, Mrows, Cc, Cc);

  // 4) recurrent scan (writes x and final state)
  rwkv_scan_kernel<<<Bc * Hc, 256, 0, stream>>>(
      rbf, kbf, vbf, state0, tdecay, tfaaaa, xbf, stateF);

  // 5) groupnorm * gate
  ln_gate_kernel<<<(Mrows * Hc) / 256, 256, 0, stream>>>(xbf, gbf, lnw, lnb, ybf);

  // 6) output projection (f32 result)
  gemm_bf16_kernel<2><<<ggrid, 256, 0, stream>>>(ybf, wob, (void*)outF, Mrows, Cc, Cc);
}
